// LSTM_CRF_51522427682872
// MI455X (gfx1250) — matrix operand order repaired and checked by reading, not compiled, not run on hardware
//
#include <hip/hip_runtime.h>

// ---------------------------------------------------------------------------
// BiLSTM-CRF loss for MI455X (gfx1250, wave32, WMMA).
// Sizes fixed by the reference: V=50000, E=300, H=256, B=64, S=256, C=19.
//
// Structure:
//   1) pack weights into WMMA B-fragment layout (bf16), gates interleaved
//   2) embed -> bf16 activations
//   3) ingemm: chip-wide WMMA GEMM, G = x @ W_ih^T  (f16, C-fragment layout)
//   4) recur:  persistent per-direction kernel, g = G[t] + h @ W_hh^T (K=256)
//      with double-buffered LDS hidden state (one barrier per step)
//   5) feats -> CRF forward + gold -> mean
// ---------------------------------------------------------------------------

typedef __attribute__((ext_vector_type(16))) __bf16 v16bf;
typedef __attribute__((ext_vector_type(8)))  float  v8f;

union ABu { v16bf v; uint4 q[2]; };
union Gp  { uint4 q; _Float16 h[8]; };

__device__ __forceinline__ unsigned short f2bf(float f) {
  union { float f; unsigned int u; } x; x.f = f;
  unsigned int r = x.u + 0x7FFFu + ((x.u >> 16) & 1u);   // round-to-nearest-even
  return (unsigned short)(r >> 16);
}
__device__ __forceinline__ float bf2f(unsigned short h) {
  union { unsigned int u; float f; } x; x.u = ((unsigned int)h) << 16;
  return x.f;
}
__device__ __forceinline__ float sigm(float x) {
  return 1.0f / (1.0f + __expf(-x));
}

// ---------------------------------------------------------------------------
// Pack weights into WMMA B-fragment layout, bf16.
// Packed column p = 4*j + gate (gate order i,f,g,o), orig row = gate*256 + j.
// Fragment layout (16-bit B, 32x16 tile): lane<16: K = 0..7 (elems 0-7), 16..23 (elems 8-15),
// N = lane; lane>=16: K = 8..15 (elems 0-7), 24..31 (elems 8-15), N = lane-16.
// KX = KT*32 & Ein<=KX  -> packs w_ih rows (k>=Ein zero-padded).
// KX = 0                -> packs w_hh rows (k = 0..KT*32-1 -> w_hh k).
// ---------------------------------------------------------------------------
__global__ void pack_w_kernel(const float* __restrict__ w_ih,
                              const float* __restrict__ w_hh,
                              int Ein, int KX, int KT,
                              unsigned short* __restrict__ wpack,
                              int total) {
  int idx = blockIdx.x * blockDim.x + threadIdx.x;
  if (idx >= total) return;
  int e    = idx & 15;
  int lane = (idx >> 4) & 31;
  int rest = idx >> 9;
  int kk   = rest % KT; rest /= KT;
  int nt   = rest % 64;
  int d    = rest / 64;
  int i = e >> 1, rbit = e & 1;
  int k = kk * 32 + ((lane < 16) ? 0 : 8) + (e & 7) + ((e & 8) ? 16 : 0);
  int p = nt * 16 + (lane & 15);
  int j = p >> 2, g = p & 3;
  int row = g * 256 + j;
  float v = 0.0f;
  if (k < KX) {
    if (k < Ein) v = w_ih[((size_t)d * 1024 + row) * Ein + k];
  } else {
    int kh = k - KX;
    v = w_hh[((size_t)d * 1024 + row) * 256 + kh];
  }
  wpack[idx] = f2bf(v);
}

__global__ void pack_b_kernel(const float* __restrict__ bi0, const float* __restrict__ bh0,
                              const float* __restrict__ bi1, const float* __restrict__ bh1,
                              float* __restrict__ bws) {
  int idx = blockIdx.x * blockDim.x + threadIdx.x;
  if (idx >= 4096) return;
  int p = idx & 1023;
  int d = (idx >> 10) & 1;
  int l = idx >> 11;
  int j = p >> 2, g = p & 3;
  int row = g * 256 + j;
  const float* bi = l ? bi1 : bi0;
  const float* bh = l ? bh1 : bh0;
  bws[idx] = bi[d * 1024 + row] + bh[d * 1024 + row];
}

// x_bf16[t][b][k] (k padded 300->320, zeros)
__global__ void embed_kernel(const int* __restrict__ sent,
                             const float* __restrict__ emb,
                             unsigned short* __restrict__ xbf) {
  size_t idx = (size_t)blockIdx.x * blockDim.x + threadIdx.x;
  const size_t total = 256ull * 64 * 320;
  if (idx >= total) return;
  int k = (int)(idx % 320);
  size_t tb = idx / 320;
  int b = (int)(tb % 64);
  int t = (int)(tb / 64);
  float v = 0.0f;
  if (k < 300) {
    int tok = sent[b * 256 + t];
    v = emb[(size_t)tok * 300 + k];
  }
  xbf[idx] = f2bf(v);
}

// ---------------------------------------------------------------------------
// Chip-wide input-projection GEMM: G = x @ W_ih^T (per direction).
// One 16x16 output tile per wave; result stored f16 in C-fragment layout:
//   gpre[((dir*1024 + rt)*64 + nt)*256 + lane*8 + r]
// so the recurrent kernel can seed its WMMA accumulators with one b128 load.
// Rows r = t*64+b are tiled as rt = t*4 + m (16 rows per tile, same t).
// ---------------------------------------------------------------------------
template <int KXT, int XS>
__global__ void __launch_bounds__(256)
ingemm_kernel(const unsigned short* __restrict__ xin,   // (16384, XS) bf16
              const unsigned short* __restrict__ wx,    // (2, 64, KXT, 512) bf16
              unsigned short* __restrict__ gpre) {      // f16 tiles
  const int tid  = threadIdx.x;
  const int wid  = tid >> 5;
  const int lane = tid & 31;
  const int lh   = lane & 15;
  const bool hi  = lane >= 16;
  const int id   = blockIdx.x * 8 + wid;
  const int nt   = id & 63;
  const int rt   = (id >> 6) & 1023;
  const int dirr = id >> 16;

  const unsigned short* wb   = wx + (((size_t)dirr * 64 + nt) * KXT) * 512 + lane * 16;
  const unsigned short* xrow = xin + (size_t)(rt * 16 + lh) * XS + (hi ? 8 : 0);

  v8f acc;
#pragma unroll
  for (int r = 0; r < 8; ++r) acc[r] = 0.0f;

#pragma unroll 2
  for (int kk = 0; kk < KXT; ++kk) {
    ABu a, b;
    b.q[0] = *(const uint4*)(wb + kk * 512);
    b.q[1] = *(const uint4*)(wb + kk * 512 + 8);
    a.q[0] = *(const uint4*)(xrow + kk * 32);
    a.q[1] = *(const uint4*)(xrow + kk * 32 + 16);
    acc = __builtin_amdgcn_wmma_f32_16x16x32_bf16(
        false, a.v, false, b.v, (short)0, acc, false, false);
  }

  Gp o;
#pragma unroll
  for (int r = 0; r < 8; ++r) o.h[r] = (_Float16)acc[r];
  *(uint4*)(gpre + (((size_t)dirr * 1024 + rt) * 64 + nt) * 256 + lane * 8) = o.q;
}

// ---------------------------------------------------------------------------
// Persistent per-direction recurrence: g = G[t] + h @ W_hh^T (K = 256).
// blockIdx.x = direction. 1024 threads = 32 waves; wave w owns n-tiles
// {2w,2w+1} == hidden units 8w..8w+7 (4 gates each), all 4 M-tiles (batch).
// Hidden state double-buffered in LDS (bf16): step s reads h_sh[s&1] and
// writes h_sh[(s&1)^1]; a single end-of-step barrier orders buffer reuse.
// ---------------------------------------------------------------------------
__global__ void __launch_bounds__(1024, 1)
recur_kernel(const unsigned short* __restrict__ gpre,   // (2,1024,64,256) f16
             const unsigned short* __restrict__ whh,    // (2,64,8,512) bf16
             const float* __restrict__ bias,            // (2,1024)
             unsigned short* __restrict__ out) {        // (S,B,512) bf16
  __shared__ __align__(16) unsigned short h_sh[2][64 * 256];  // 2 x 32 KB
  __shared__ __align__(16) float stage[32 * 256];             // 32 KB

  const int tid  = threadIdx.x;
  const int wid  = tid >> 5;
  const int lane = tid & 31;
  const int lh   = lane & 15;
  const bool hi  = lane >= 16;
  const int dir  = blockIdx.x;
  const int aoff = hi ? 8 : 0;

  const float* bp = bias + dir * 1024;
  float* st = stage + wid * 256;

  const unsigned short* wpn0  = whh + ((size_t)dir * 64 + 2 * wid) * 8 * 512 + lane * 16;
  const unsigned short* wpn1  = wpn0 + 8 * 512;
  const unsigned short* gbase = gpre + (size_t)dir * 1024 * 64 * 256
                              + (size_t)(2 * wid) * 256 + lane * 8;

  for (int i = tid; i < 64 * 256; i += 1024) h_sh[0][i] = 0;

  float cst[4][2][2];
#pragma unroll
  for (int m = 0; m < 4; ++m)
#pragma unroll
    for (int n = 0; n < 2; ++n) { cst[m][n][0] = 0.f; cst[m][n][1] = 0.f; }

  __syncthreads();

  for (int step = 0; step < 256; ++step) {
    const int t = dir ? (255 - step) : step;
    const unsigned short* hrd = h_sh[step & 1];
    unsigned short*       hwr = h_sh[(step & 1) ^ 1];

    // Seed accumulators with the precomputed input projection (f16 -> f32).
    v8f acc[4][2];
#pragma unroll
    for (int m = 0; m < 4; ++m)
#pragma unroll
      for (int n = 0; n < 2; ++n) {
        Gp gi;
        gi.q = *(const uint4*)(gbase + ((size_t)(t * 4 + m) * 64 + n) * 256);
#pragma unroll
        for (int r = 0; r < 8; ++r) acc[m][n][r] = (float)gi.h[r];
      }

    // h-part: A fragments from LDS hidden state, B from packed W_hh.
#pragma unroll
    for (int kk = 0; kk < 8; ++kk) {
      const int kh = kk * 32 + aoff;
      ABu b0, b1;
      b0.q[0] = *(const uint4*)(wpn0 + kk * 512);
      b0.q[1] = *(const uint4*)(wpn0 + kk * 512 + 8);
      b1.q[0] = *(const uint4*)(wpn1 + kk * 512);
      b1.q[1] = *(const uint4*)(wpn1 + kk * 512 + 8);
#pragma unroll
      for (int m = 0; m < 4; ++m) {
        ABu af;
        const unsigned short* p = hrd + (m * 16 + lh) * 256 + kh;
        af.q[0] = *(const uint4*)(p);
        af.q[1] = *(const uint4*)(p + 16);
        acc[m][0] = __builtin_amdgcn_wmma_f32_16x16x32_bf16(
            false, af.v, false, b0.v, (short)0, acc[m][0], false, false);
        acc[m][1] = __builtin_amdgcn_wmma_f32_16x16x32_bf16(
            false, af.v, false, b1.v, (short)0, acc[m][1], false, false);
      }
    }

    // Stage each tile to per-wave LDS, cell update, write h to the other
    // LDS buffer (safe: barrier below orders buffer reuse) and to global.
#pragma unroll
    for (int m = 0; m < 4; ++m) {
#pragma unroll
      for (int n = 0; n < 2; ++n) {
#pragma unroll
        for (int r = 0; r < 8; ++r) {
          const int row = hi ? (8 + r) : r;
          st[row * 16 + lh] = acc[m][n][r];
        }
#pragma unroll
        for (int s = 0; s < 2; ++s) {
          const int q  = lane + 32 * s;
          const int bl = q >> 2;
          const int jl = q & 3;
          const float4 g  = *reinterpret_cast<const float4*>(&st[bl * 16 + jl * 4]);
          const int jg = wid * 8 + n * 4 + jl;
          const float4 bb = *reinterpret_cast<const float4*>(&bp[jg * 4]);
          const float iv = sigm(g.x + bb.x);
          const float fv = sigm(g.y + bb.y);
          const float gv = tanhf(g.z + bb.z);
          const float ov = sigm(g.w + bb.w);
          float c = fv * cst[m][n][s] + iv * gv;
          cst[m][n][s] = c;
          const float h = ov * tanhf(c);
          const unsigned short hb = f2bf(h);
          const int b = m * 16 + bl;
          hwr[b * 256 + jg] = hb;
          out[((size_t)t * 64 + b) * 512 + dir * 256 + jg] = hb;
        }
      }
    }

    __syncthreads();   // h_t complete before any wave starts step t+1
  }
}

// feats[b][t][c] = out1[t][b][:] . lin_w[c][:] + lin_b[c]
__global__ void feats_kernel(const unsigned short* __restrict__ out1,
                             const float* __restrict__ lw,
                             const float* __restrict__ lb,
                             float* __restrict__ feats) {
  int idx = blockIdx.x * blockDim.x + threadIdx.x;
  if (idx >= 64 * 256 * 19) return;
  int c = idx % 19;
  int rest = idx / 19;
  int t = rest % 256;
  int b = rest / 256;
  const unsigned short* o = out1 + ((size_t)t * 64 + b) * 512;
  const float* w = lw + (size_t)c * 512;
  float acc = lb[c];
#pragma unroll 8
  for (int k = 0; k < 512; ++k) acc += bf2f(o[k]) * w[k];
  feats[idx] = acc;
}

// One wave per batch row: CRF forward (alpha per lane) + gold score.
__global__ void crf_kernel(const float* __restrict__ feats,
                           const int* __restrict__ labels,
                           const int* __restrict__ lens,
                           const float* __restrict__ T,
                           float* __restrict__ diffs) {
  __shared__ float Ts[19 * 19];
  const int b = blockIdx.x;
  const int lane = threadIdx.x;
  for (int i = lane; i < 361; i += 32) Ts[i] = T[i];
  __syncthreads();

  const int len = lens[b];
  const int c = lane;
  const bool act = c < 19;
  float alpha = (c == 17) ? 0.0f : -10000.0f;   // START=17

  for (int t = 0; t < 256; ++t) {
    const float feat = act ? feats[((size_t)b * 256 + t) * 19 + c] : 0.0f;
    float m = -3.0e38f, ssum = 0.0f;
#pragma unroll
    for (int p = 0; p < 19; ++p) {
      const float ap = __shfl(alpha, p, 32);
      const float v = ap + (act ? Ts[c * 19 + p] : 0.0f);
      if (v > m) { ssum = ssum * __expf(m - v) + 1.0f; m = v; }
      else       { ssum += __expf(v - m); }
    }
    const float an = feat + m + __logf(ssum);
    if (act && t < len) alpha = an;
  }

  // fwd = logsumexp_c(alpha[c] + T[STOP=18][c])
  float v = act ? (alpha + Ts[18 * 19 + c]) : -3.0e38f;
  float m = v;
#pragma unroll
  for (int off = 16; off; off >>= 1) m = fmaxf(m, __shfl_xor(m, off, 32));
  float e = act ? __expf(v - m) : 0.0f;
#pragma unroll
  for (int off = 16; off; off >>= 1) e += __shfl_xor(e, off, 32);
  const float fwd = m + __logf(e);

  if (lane == 0) {
    float gold = 0.0f;
    int prev = 17;                                  // lab[0] = START
    for (int i = 0; i <= 256; ++i) {
      int nxt;
      if (i < 256) {
        const int lv = labels[b * 256 + i];
        nxt = (lv > 0) ? lv : 18;                   // where(lab>0, lab, STOP)
        if (i < len) gold += feats[((size_t)b * 256 + i) * 19 + lv];
      } else {
        nxt = 18;                                   // STOP
      }
      if (i < len + 1) gold += Ts[nxt * 19 + prev]; // T[lab[i+1], lab[i]]
      prev = nxt;
    }
    diffs[b] = fwd - gold;
  }
}

__global__ void final_kernel(const float* __restrict__ diffs, float* __restrict__ out) {
  __shared__ float s[64];
  s[threadIdx.x] = diffs[threadIdx.x];
  __syncthreads();
  if (threadIdx.x == 0) {
    float acc = 0.0f;
    for (int i = 0; i < 64; ++i) acc += s[i];
    out[0] = acc * (1.0f / 64.0f);
  }
}

// ---------------------------------------------------------------------------
extern "C" void kernel_launch(void* const* d_in, const int* in_sizes, int n_in,
                              void* d_out, int out_size, void* d_ws, size_t ws_size,
                              hipStream_t stream) {
  const int*   sent   = (const int*)d_in[0];
  const int*   labels = (const int*)d_in[1];
  const int*   lens   = (const int*)d_in[2];
  const float* emb    = (const float*)d_in[3];
  const float* w_ih0  = (const float*)d_in[4];
  const float* w_hh0  = (const float*)d_in[5];
  const float* b_ih0  = (const float*)d_in[6];
  const float* b_hh0  = (const float*)d_in[7];
  const float* w_ih1  = (const float*)d_in[8];
  const float* w_hh1  = (const float*)d_in[9];
  const float* b_ih1  = (const float*)d_in[10];
  const float* b_hh1  = (const float*)d_in[11];
  const float* lin_w  = (const float*)d_in[12];
  const float* lin_b  = (const float*)d_in[13];
  const float* trans  = (const float*)d_in[14];
  (void)in_sizes; (void)n_in; (void)out_size; (void)ws_size;

  char* ws = (char*)d_ws;
  size_t off = 0;
  auto take = [&](size_t bytes) -> char* {
    char* p = ws + off;
    off = (off + bytes + 255) & ~(size_t)255;
    return p;
  };
  unsigned short* xbf   = (unsigned short*)take(256ull * 64 * 320 * 2);
  unsigned short* out0  = (unsigned short*)take(256ull * 64 * 512 * 2);
  unsigned short* out1  = (unsigned short*)take(256ull * 64 * 512 * 2);
  unsigned short* gpre  = (unsigned short*)take(2ull * 1024 * 64 * 256 * 2);  // 67 MB, reused per layer
  unsigned short* wx0   = (unsigned short*)take(2ull * 64 * 10 * 512 * 2);
  unsigned short* wx1   = (unsigned short*)take(2ull * 64 * 16 * 512 * 2);
  unsigned short* whh0  = (unsigned short*)take(2ull * 64 * 8 * 512 * 2);
  unsigned short* whh1  = (unsigned short*)take(2ull * 64 * 8 * 512 * 2);
  float*          biasw = (float*)take(4096 * 4);
  float*          featw = (float*)take(64ull * 256 * 19 * 4);
  float*          diffs = (float*)take(64 * 4);

  // Weight / bias packing.
  pack_w_kernel<<<2560, 256, 0, stream>>>(w_ih0, w_hh0, 300, 320, 10, wx0, 655360);
  pack_w_kernel<<<4096, 256, 0, stream>>>(w_ih1, w_hh1, 512, 512, 16, wx1, 1048576);
  pack_w_kernel<<<2048, 256, 0, stream>>>(w_hh0, w_hh0, 0, 0, 8, whh0, 524288);
  pack_w_kernel<<<2048, 256, 0, stream>>>(w_hh1, w_hh1, 0, 0, 8, whh1, 524288);
  pack_b_kernel<<<16, 256, 0, stream>>>(b_ih0, b_hh0, b_ih1, b_hh1, biasw);
  embed_kernel<<<20480, 256, 0, stream>>>(sent, emb, xbf);

  // Layer 0: chip-wide input projection, then recurrence.
  ingemm_kernel<10, 320><<<16384, 256, 0, stream>>>(xbf, wx0, gpre);
  recur_kernel<<<2, 1024, 0, stream>>>(gpre, whh0, biasw, out0);

  // Layer 1.
  ingemm_kernel<16, 512><<<16384, 256, 0, stream>>>(out0, wx1, gpre);
  recur_kernel<<<2, 1024, 0, stream>>>(gpre, whh1, biasw + 2048, out1);

  feats_kernel<<<(64 * 256 * 19 + 255) / 256, 256, 0, stream>>>(out1, lin_w, lin_b, featw);
  crf_kernel<<<64, 32, 0, stream>>>(featw, labels, lens, trans, diffs);
  final_kernel<<<1, 64, 0, stream>>>(diffs, (float*)d_out);
}
